// Transformer_13735305413384
// MI455X (gfx1250) — compile-verified, matrix operand order checked by reading, not run on hardware
//
#include <hip/hip_runtime.h>
#include <hip/hip_bf16.h>

// ---------------------------------------------------------------------------
// Transformer block forward for MI455X (gfx1250), wave32 + WMMA bf16.
//   B=512, S=50, D=2048, H=16, DH=128
// GEMMs in bf16 (v_wmma_f32_16x16x32_bf16, fp32 accumulate); LN / softmax /
// residuals in fp32. CDNA5 data movement: TDM tensor_load_to_lds for the
// attention activation tile, async global->LDS double buffering in the MLP.
// ---------------------------------------------------------------------------

#define BATCH   512
#define SEQL    50
#define DMODEL  2048
#define NHEAD   16
#define DH      128
#define NTOK    (BATCH * SEQL)        // 25600
#define SPAD    64                     // SEQ padded to 4 WMMA tiles

#ifndef __has_builtin
#define __has_builtin(x) 0
#endif
#define HAS_ASYNC_LDS __has_builtin(__builtin_amdgcn_global_load_async_to_lds_b128)
#define HAS_TDM       __has_builtin(__builtin_amdgcn_tensor_load_to_lds)

typedef __attribute__((ext_vector_type(16))) __bf16 v16bf;
typedef __attribute__((ext_vector_type(8)))  float  v8f;
typedef __attribute__((ext_vector_type(4)))  unsigned int v4u;
typedef __attribute__((ext_vector_type(8)))  int    v8i;
typedef __attribute__((ext_vector_type(4)))  int    v4i;

#if HAS_ASYNC_LDS
typedef __attribute__((address_space(1))) v4i* gv4ip;   // global v4i*
typedef __attribute__((address_space(3))) v4i* lv4ip;   // LDS v4i*
#endif

union FragU { v16bf v; uint4 q[2]; };

// Load one 16x32 bf16 A/B fragment from a row-major-over-K array.
// Lane layout (ISA 7.12.2, 16-bit 16x32): lane L<16 -> row L, K in {0..7,16..23};
// lane L>=16 -> row L-16, K in {8..15,24..31}. Two 16-byte LDS loads per lane.
__device__ __forceinline__ v16bf ld_frag(const __bf16* base, int row, int ld,
                                         int k0, int lane) {
    const int khalf = (lane & 16) ? 8 : 0;
    const __bf16* p = base + row * ld + k0 + khalf;
    FragU f;
    f.q[0] = *(const uint4*)(p);        // K = k0+khalf   .. +7
    f.q[1] = *(const uint4*)(p + 16);   // K = k0+khalf+16 .. +23
    return f.v;
}

__device__ __forceinline__ v8f wmma_bf16(v16bf a, v16bf b, v8f c) {
    return __builtin_amdgcn_wmma_f32_16x16x32_bf16(
        false, a, false, b, (short)0, c, false, false);
}

__device__ __forceinline__ void wait_async0() {
#if HAS_ASYNC_LDS
#if __has_builtin(__builtin_amdgcn_s_wait_asynccnt)
    __builtin_amdgcn_s_wait_asynccnt(0);
#else
    asm volatile("s_wait_asynccnt 0x0" ::: "memory");
#endif
#endif
}

// ---------------------------------------------------------------------------
// Kernel 1/4: LayerNorm (fp32 in) -> bf16 out.  One 256-thread block per token.
// ---------------------------------------------------------------------------
__global__ __launch_bounds__(256) void ln_to_bf16(
    const float* __restrict__ x, const float* __restrict__ gamma,
    const float* __restrict__ beta, __bf16* __restrict__ out) {
    __shared__ float red[256];
    const int t   = blockIdx.x;
    const int tid = threadIdx.x;
    const float* xr = x + (size_t)t * DMODEL;

    float v[8];
    float s = 0.f;
#pragma unroll
    for (int i = 0; i < 8; ++i) { v[i] = xr[tid + 256 * i]; s += v[i]; }

    red[tid] = s; __syncthreads();
    for (int o = 128; o > 0; o >>= 1) {
        if (tid < o) red[tid] += red[tid + o];
        __syncthreads();
    }
    const float mu = red[0] * (1.0f / (float)DMODEL);
    __syncthreads();

    float sq = 0.f;
#pragma unroll
    for (int i = 0; i < 8; ++i) { float d = v[i] - mu; sq += d * d; }
    red[tid] = sq; __syncthreads();
    for (int o = 128; o > 0; o >>= 1) {
        if (tid < o) red[tid] += red[tid + o];
        __syncthreads();
    }
    const float var  = red[0] * (1.0f / (float)DMODEL);
    const float rstd = rsqrtf(var + 1e-5f);

#pragma unroll
    for (int i = 0; i < 8; ++i) {
        const int c = tid + 256 * i;
        out[(size_t)t * DMODEL + c] =
            (__bf16)((v[i] - mu) * rstd * gamma[c] + beta[c]);
    }
}

// ---------------------------------------------------------------------------
// Kernel 2: fp32 -> bf16 bulk convert (for Wm).
// ---------------------------------------------------------------------------
__global__ __launch_bounds__(256) void f32_to_bf16(
    const float* __restrict__ in, __bf16* __restrict__ out, int n) {
    int i = (blockIdx.x * 256 + threadIdx.x) * 4;
    if (i + 3 < n) {
        float4 f = *(const float4*)(in + i);
        out[i + 0] = (__bf16)f.x;
        out[i + 1] = (__bf16)f.y;
        out[i + 2] = (__bf16)f.z;
        out[i + 3] = (__bf16)f.w;
    }
}

// ---------------------------------------------------------------------------
// Kernel 3: fused QKV + attention + first residual, one WG per (batch, head).
// 128 threads = 4 waves; wave w owns query rows [16w, 16w+16).
// Writes d_out = x + attn (x2).
// ---------------------------------------------------------------------------
__global__ __launch_bounds__(128) void attn_kernel(
    const __bf16* __restrict__ Xn, const float* __restrict__ x,
    const float* __restrict__ Wq, const float* __restrict__ bq,
    const float* __restrict__ Wk, const float* __restrict__ bk,
    const float* __restrict__ Wv, const float* __restrict__ bv,
    float* __restrict__ out) {
    __shared__ __align__(16) __bf16 sW [DH * DH];      // staged weight (32 KB)
    __shared__ __align__(16) __bf16 sX [SPAD * DH];    // LN'd slice   (16 KB)
    __shared__ __align__(16) __bf16 sQ [SPAD * DH];    // Q            (16 KB)
    __shared__ __align__(16) __bf16 sK [SPAD * DH];    // K            (16 KB)
    __shared__ __align__(16) __bf16 sVT[DH * SPAD];    // V transposed (16 KB)
    __shared__ __align__(16) __bf16 sP [SPAD * SPAD];  // probs        ( 8 KB)
    __shared__ float sb[DH];

    const int b    = blockIdx.x >> 4;
    const int h    = blockIdx.x & 15;
    const int tid  = threadIdx.x;
    const int lane = tid & 31;
    const int w    = tid >> 5;
    const int ln   = lane & 15;
    const int hi8  = (lane & 16) ? 8 : 0;

    // ---- stage LN'd head slice (zero-pad rows SEQL..63) ----
    const __bf16* xnrow = Xn + (size_t)b * SEQL * DMODEL + h * DH;
#if HAS_TDM
    // zero the pad rows while the DMA is in flight (disjoint LDS range)
    for (int i = tid; i < (SPAD - SEQL) * DH; i += 128)
        sX[SEQL * DH + i] = (__bf16)0.0f;
    if (tid < 32) {
        // Tensor DMA descriptor (CDNA5 ISA):
        // 2D tile: tile_dim0 = 128 bf16 elems, tile_dim1 = 50 rows,
        // tensor_dim0_stride = 2048 elems, data_size = 2 bytes.
        const unsigned long long ga = (unsigned long long)(size_t)(const void*)xnrow;
        const unsigned lds_off = (unsigned)(size_t)(void*)&sX[0];  // LDS aperture low bits
        v4u g0;
        g0.x = 1u;                                        // count=1, user mode
        g0.y = lds_off;                                   // lds_addr
        g0.z = (unsigned)(ga & 0xffffffffu);              // global_addr[31:0]
        g0.w = (unsigned)((ga >> 32) & 0x1ffffffu) | (2u << 30);  // addr[56:32], type=2
        v8i g1;
        g1[0] = (int)(1u << 16);                          // data_size=1 (2 bytes)
        g1[1] = (int)((DMODEL & 0xffffu) << 16);          // tensor_dim0 lo16 @ [63:48]
        g1[2] = (int)((DMODEL >> 16) | ((4096u & 0xffffu) << 16)); // dim0 hi / dim1 lo
        g1[3] = (int)((4096u >> 16) | (128u << 16));      // dim1 hi / tile_dim0=128
        g1[4] = (int)SEQL;                                // tile_dim1=50, tile_dim2=0
        g1[5] = (int)DMODEL;                              // tensor_dim0_stride lo32
        g1[6] = 0;                                        // stride hi / dim1_stride lo
        g1[7] = 0;
        v4i gz4 = {0, 0, 0, 0};                           // groups 2/3: 2D tensor
        v8i gz8 = {0, 0, 0, 0, 0, 0, 0, 0};
        __builtin_amdgcn_tensor_load_to_lds(g0, g1, gz4, gz4, gz8, 0);
    }
    __builtin_amdgcn_s_wait_tensorcnt(0);
#else
    for (int i = tid; i < SPAD * DH; i += 128) {
        const int s = i >> 7, d = i & 127;
        sX[i] = (s < SEQL) ? xnrow[(size_t)s * DMODEL + d] : (__bf16)0.0f;
    }
#endif

    const float* Wsrc[3] = {Wq + (size_t)h * DH * DH, Wk + (size_t)h * DH * DH,
                            Wv + (size_t)h * DH * DH};
    const float* Bsrc[3] = {bq + h * DH, bk + h * DH, bv + h * DH};

    // ---- QKV: D[s][e] = sum_d X[s][d] * W[e][d] + bias[e] ----
    for (int mat = 0; mat < 3; ++mat) {
        __syncthreads();
        const float* wsrc = Wsrc[mat];
        for (int i = tid; i < DH * DH; i += 128) sW[i] = (__bf16)wsrc[i];
        if (tid < DH) sb[tid] = Bsrc[mat][tid];
        __syncthreads();

        for (int n = 0; n < 8; ++n) {
            const float be = sb[n * 16 + ln];
            v8f acc;
#pragma unroll
            for (int r = 0; r < 8; ++r) acc[r] = be;
#pragma unroll
            for (int kk = 0; kk < 4; ++kk) {
                v16bf a  = ld_frag(sX, 16 * w + ln, DH, 32 * kk, lane);
                v16bf bb = ld_frag(sW, n * 16 + ln, DH, 32 * kk, lane);
                acc = wmma_bf16(a, bb, acc);
            }
            if (mat == 0) {
#pragma unroll
                for (int r = 0; r < 8; ++r)
                    sQ[(16 * w + r + hi8) * DH + n * 16 + ln] = (__bf16)acc[r];
            } else if (mat == 1) {
#pragma unroll
                for (int r = 0; r < 8; ++r)
                    sK[(16 * w + r + hi8) * DH + n * 16 + ln] = (__bf16)acc[r];
            } else {
#pragma unroll
                for (int r = 0; r < 8; ++r)
                    sVT[(n * 16 + ln) * SPAD + (16 * w + r + hi8)] = (__bf16)acc[r];
            }
        }
    }
    __syncthreads();

    // ---- scores[s][t] = sqrt(D) * sum_e Q[s][e] K[t][e] ----
    v8f accS[4];
#pragma unroll
    for (int tn = 0; tn < 4; ++tn)
#pragma unroll
        for (int r = 0; r < 8; ++r) accS[tn][r] = 0.f;

    for (int kk = 0; kk < 8; ++kk) {
        v16bf a = ld_frag(sQ, 16 * w + ln, DH, 32 * kk, lane);
#pragma unroll
        for (int tn = 0; tn < 4; ++tn) {
            v16bf bb = ld_frag(sK, tn * 16 + ln, DH, 32 * kk, lane);
            accS[tn] = wmma_bf16(a, bb, accS[tn]);
        }
    }

    // ---- softmax per row (C-fragment: row r+hi8, col = tn*16+ln) ----
    const float SCALE = 45.254833995939045f;  // sqrt(2048)
#pragma unroll
    for (int r = 0; r < 8; ++r) {
        float vals[4];
        float m = -3.0e38f;
#pragma unroll
        for (int tn = 0; tn < 4; ++tn) {
            float vv = accS[tn][r] * SCALE;
            if (tn * 16 + ln >= SEQL) vv = -3.0e38f;
            vals[tn] = vv;
            m = fmaxf(m, vv);
        }
        for (int o = 8; o > 0; o >>= 1) m = fmaxf(m, __shfl_xor(m, o, 16));
        float ssum = 0.f;
#pragma unroll
        for (int tn = 0; tn < 4; ++tn) {
            float e = __expf(vals[tn] - m);
            vals[tn] = e;
            ssum += e;
        }
        for (int o = 8; o > 0; o >>= 1) ssum += __shfl_xor(ssum, o, 16);
        const float inv = 1.0f / ssum;
        const int row = 16 * w + r + hi8;
#pragma unroll
        for (int tn = 0; tn < 4; ++tn)
            sP[row * SPAD + tn * 16 + ln] = (__bf16)(vals[tn] * inv);
    }
    __syncthreads();

    // ---- attn[s][e] = sum_t P[s][t] V[t][e];  out = x + attn ----
    const float* xres = x   + (size_t)b * SEQL * DMODEL + h * DH;
    float*       orow = out + (size_t)b * SEQL * DMODEL + h * DH;
    for (int n = 0; n < 8; ++n) {
        v8f acc;
#pragma unroll
        for (int r = 0; r < 8; ++r) acc[r] = 0.f;
#pragma unroll
        for (int kk = 0; kk < 2; ++kk) {
            v16bf a  = ld_frag(sP,  16 * w + ln, SPAD, 32 * kk, lane);
            v16bf bb = ld_frag(sVT, n * 16 + ln, SPAD, 32 * kk, lane);
            acc = wmma_bf16(a, bb, acc);
        }
#pragma unroll
        for (int r = 0; r < 8; ++r) {
            const int srow = 16 * w + r + hi8;
            if (srow < SEQL) {
                const int col = n * 16 + ln;
                orow[(size_t)srow * DMODEL + col] =
                    xres[(size_t)srow * DMODEL + col] + acc[r];
            }
        }
    }
}

// ---------------------------------------------------------------------------
// Kernel 5: MLP GEMM  out[t][j] += sum_i Xn2[t][i] * Wm[j][i] + bm[j]
// 256 threads = 8 waves; WG tile 128x128; K staged 64 at a time, double
// buffered in LDS via async global->LDS copies (ASYNCcnt).
// ---------------------------------------------------------------------------
#define KB_STEPS (DMODEL / 64)   // 32

__device__ __forceinline__ void mlp_stage(const __bf16* __restrict__ Xn,
                                          const __bf16* __restrict__ Wmb,
                                          __bf16* sA, __bf16* sB,
                                          int m0, int n0, int kb, int tid) {
#pragma unroll
    for (int i = 0; i < 4; ++i) {
        const int c   = tid + 256 * i;       // 1024 16-byte chunks total
        const int row = c >> 3;
        const int k8  = (c & 7) * 8;
        const __bf16* ga = Xn  + (size_t)(m0 + row) * DMODEL + kb * 64 + k8;
        const __bf16* gb = Wmb + (size_t)(n0 + row) * DMODEL + kb * 64 + k8;
        __bf16* la = sA + row * 64 + k8;
        __bf16* lb = sB + row * 64 + k8;
#if HAS_ASYNC_LDS
        __builtin_amdgcn_global_load_async_to_lds_b128(
            (gv4ip)(v4i*)(void*)ga, (lv4ip)(v4i*)(void*)la, 0, 0);
        __builtin_amdgcn_global_load_async_to_lds_b128(
            (gv4ip)(v4i*)(void*)gb, (lv4ip)(v4i*)(void*)lb, 0, 0);
#else
        *(uint4*)la = *(const uint4*)ga;
        *(uint4*)lb = *(const uint4*)gb;
#endif
    }
}

__global__ __launch_bounds__(256) void mlp_kernel(
    const __bf16* __restrict__ Xn, const __bf16* __restrict__ Wmb,
    const float* __restrict__ bm, float* __restrict__ out) {
    __shared__ __align__(16) __bf16 sA[2 * 128 * 64];   // 32 KB double-buffered
    __shared__ __align__(16) __bf16 sB[2 * 128 * 64];   // 32 KB double-buffered

    const int m0   = blockIdx.x * 128;
    const int n0   = blockIdx.y * 128;
    const int tid  = threadIdx.x;
    const int lane = tid & 31;
    const int w    = tid >> 5;
    const int ln   = lane & 15;
    const int hi8  = (lane & 16) ? 8 : 0;

    v8f acc[8];
#pragma unroll
    for (int n = 0; n < 8; ++n)
#pragma unroll
        for (int r = 0; r < 8; ++r) acc[n][r] = 0.f;

    mlp_stage(Xn, Wmb, sA, sB, m0, n0, 0, tid);

    for (int kb = 0; kb < KB_STEPS; ++kb) {
        wait_async0();          // my async copies for the current buffer landed
        __syncthreads();        // everyone's copies landed / prior reads done
        if (kb + 1 < KB_STEPS) {
            const int nb = (kb + 1) & 1;
            mlp_stage(Xn, Wmb, sA + nb * (128 * 64), sB + nb * (128 * 64),
                      m0, n0, kb + 1, tid);
        }
        const __bf16* curA = sA + (kb & 1) * (128 * 64);
        const __bf16* curB = sB + (kb & 1) * (128 * 64);
#pragma unroll
        for (int kk = 0; kk < 2; ++kk) {
            v16bf a = ld_frag(curA, 16 * w + ln, 64, 32 * kk, lane);
#pragma unroll
            for (int n = 0; n < 8; ++n) {
                v16bf bb = ld_frag(curB, n * 16 + ln, 64, 32 * kk, lane);
                acc[n] = wmma_bf16(a, bb, acc[n]);
            }
        }
    }

    // epilogue: out += acc + bias (d_out already holds x2)
#pragma unroll
    for (int n = 0; n < 8; ++n) {
        const int col = n0 + n * 16 + ln;
        const float bias = bm[col];
#pragma unroll
        for (int r = 0; r < 8; ++r) {
            const size_t idx = (size_t)(m0 + 16 * w + r + hi8) * DMODEL + col;
            out[idx] = out[idx] + acc[n][r] + bias;
        }
    }
}

// ---------------------------------------------------------------------------
extern "C" void kernel_launch(void* const* d_in, const int* in_sizes, int n_in,
                              void* d_out, int out_size, void* d_ws,
                              size_t ws_size, hipStream_t stream) {
    (void)in_sizes; (void)n_in; (void)out_size; (void)ws_size;

    const float* x      = (const float*)d_in[0];
    const float* gamma1 = (const float*)d_in[1];
    const float* beta1  = (const float*)d_in[2];
    const float* gamma2 = (const float*)d_in[3];
    const float* beta2  = (const float*)d_in[4];
    const float* Wq     = (const float*)d_in[5];
    const float* bq     = (const float*)d_in[6];
    const float* Wk     = (const float*)d_in[7];
    const float* bk     = (const float*)d_in[8];
    const float* Wv     = (const float*)d_in[9];
    const float* bv     = (const float*)d_in[10];
    const float* Wm     = (const float*)d_in[11];
    const float* bm     = (const float*)d_in[12];
    float* out = (float*)d_out;

    // workspace layout: [Xn bf16: NTOK*DMODEL] [Wm bf16: DMODEL*DMODEL]
    __bf16* Xn  = (__bf16*)d_ws;
    __bf16* Wmb = (__bf16*)((char*)d_ws + (size_t)NTOK * DMODEL * 2);

    // 1) LN1 -> bf16
    ln_to_bf16<<<NTOK, 256, 0, stream>>>(x, gamma1, beta1, Xn);
    // 2) Wm -> bf16 (L2-resident thereafter)
    f32_to_bf16<<<(DMODEL * DMODEL) / (256 * 4), 256, 0, stream>>>(
        Wm, Wmb, DMODEL * DMODEL);
    // 3) QKV + attention + residual -> d_out = x2
    attn_kernel<<<BATCH * NHEAD, 128, 0, stream>>>(Xn, x, Wq, bq, Wk, bk, Wv,
                                                   bv, out);
    // 4) LN2 on x2 -> bf16 (reuse Xn)
    ln_to_bf16<<<NTOK, 256, 0, stream>>>(out, gamma2, beta2, Xn);
    // 5) MLP: out += Xn2 @ Wm^T + bm
    mlp_kernel<<<dim3(NTOK / 128, DMODEL / 128), 256, 0, stream>>>(Xn, Wmb, bm,
                                                                   out);
}
